// AgentAttention_19061064860370
// MI455X (gfx1250) — matrix operand order repaired and checked by reading, not compiled, not run on hardware
//
#include <hip/hip_runtime.h>

typedef __attribute__((ext_vector_type(16))) _Float16 v16h;
typedef __attribute__((ext_vector_type(8)))  _Float16 v8h;
typedef __attribute__((ext_vector_type(4)))  _Float16 v4h;
typedef __attribute__((ext_vector_type(8)))  float    v8f;
typedef __attribute__((ext_vector_type(4)))  unsigned int u32x4;
typedef __attribute__((ext_vector_type(8)))  int i32x8;
typedef __attribute__((ext_vector_type(4)))  int i32x4;

#define BATCH 16
#define CH    512
#define NTOK  3136
#define HEADS 8
#define HD    64
#define AGENT 49
#define MTOT  (BATCH*NTOK)   /* 50176 */
#define SCALE 0.125f         /* hd^-0.5 */

union V16u { v16h v; v8h h[2]; };

#if defined(__gfx1250__) && __has_builtin(__builtin_amdgcn_tensor_load_to_lds)
#define USE_TDM 1
#else
#define USE_TDM 0
#endif

#if USE_TDM
// ---------------------------------------------------------------------------
// TDM: DMA a 128x32 f16 tile (row stride 512 halves in memory) into LDS with
// row padding 64B->80B (pad_interval=16 dwords, pad_amount=4 dwords) so the
// LDS image has LD=40 halves. D# built per CDNA5 ISA 8.3-8.6. One issue per
// wave (EXEC ignored); tracked by TENSORcnt.
// ---------------------------------------------------------------------------
__device__ __forceinline__ void tdm_tile_128x32_f16(const _Float16* g, unsigned ldsOff)
{
  unsigned long long ga = (unsigned long long)(uintptr_t)g;
  u32x4 g0;
  g0[0] = 1u;                                               // count=1 (valid)
  g0[1] = ldsOff;                                           // lds_addr (bytes)
  g0[2] = (unsigned)ga;                                     // global_addr[31:0]
  g0[3] = ((unsigned)(ga >> 32) & 0x01FFFFFFu) | 0x80000000u; // ga[56:32] | type=2
  i32x8 g1;
  g1[0] = (1 << 16) | (1 << 20) | (3 << 22) | (3 << 25);    // data_size=2B, pad 4dw/16dw
  g1[1] = (int)(512u << 16);                                // tensor_dim0 = 512
  g1[2] = 0;                                                // tensor_dim1 lo16 = 0
  g1[3] = 16 | (32 << 16);                                  // tensor_dim1 = 1<<20, tile_dim0=32
  g1[4] = 128;                                              // tile_dim1=128, tile_dim2=0
  g1[5] = 512;                                              // tensor_dim0_stride = 512
  g1[6] = 0;
  g1[7] = 0;
  i32x4 z4 = {0, 0, 0, 0};
#if __clang_major__ >= 23
  i32x8 z8 = {0, 0, 0, 0, 0, 0, 0, 0};
  __builtin_amdgcn_tensor_load_to_lds(g0, g1, z4, z4, z8, 0);
#else
  __builtin_amdgcn_tensor_load_to_lds(g0, g1, z4, z4, 0);
#endif
}
#endif

// ---------------------------------------------------------------------------
// WMMA GEMM: C[M x 512] = A[M x 512] * Bw[512 x 512]^T (+bias), f16 in /
// fp32 out, v_wmma_f32_16x16x32_f16. Block tile 128x128, K-step 32,
// 256 thr = 8 waves, wave tile 32x64. TDM double-buffered LDS staging.
// ---------------------------------------------------------------------------
template <bool HAS_BIAS>
__global__ __launch_bounds__(256) void gemm512_wmma(
    const _Float16* __restrict__ A, const _Float16* __restrict__ Bw,
    float* __restrict__ Cmat, const float* __restrict__ bias)
{
  constexpr int K = 512, N = 512, LD = 40;   // padded LDS row (16B aligned)
  __shared__ __align__(16) _Float16 sA[2][128 * LD];
  __shared__ __align__(16) _Float16 sB[2][128 * LD];

  const int tid  = threadIdx.x;
  const int lane = tid & 31;
  const int wid  = tid >> 5;
  const int wm   = wid & 3;   // 4 waves along M
  const int wn   = wid >> 2;  // 2 waves along N
  const long rowBase = (long)blockIdx.x * 128;
  const long colBase = (long)blockIdx.y * 128;

  const v8f vzero = {0.f,0.f,0.f,0.f,0.f,0.f,0.f,0.f};
  v8f acc[2][4];
  for (int i = 0; i < 2; ++i)
    for (int j = 0; j < 4; ++j) acc[i][j] = vzero;

  const int r16   = lane & 15;
  const int khalf = lane >> 4;
  const _Float16* Arow = A  + rowBase * K;
  const _Float16* Brow = Bw + colBase * K;

#if USE_TDM
  if (wid == 0) {
    tdm_tile_128x32_f16(Arow, (unsigned)(uintptr_t)&sA[0][0]);
    tdm_tile_128x32_f16(Brow, (unsigned)(uintptr_t)&sB[0][0]);
  }
#endif

  for (int kk = 0; kk < K / 32; ++kk) {
    const int buf = kk & 1;
#if USE_TDM
    if (wid == 0) {
      if (kk + 1 < K / 32) {   // prefetch next tiles into the other buffer
        tdm_tile_128x32_f16(Arow + (kk + 1) * 32, (unsigned)(uintptr_t)&sA[buf ^ 1][0]);
        tdm_tile_128x32_f16(Brow + (kk + 1) * 32, (unsigned)(uintptr_t)&sB[buf ^ 1][0]);
        __builtin_amdgcn_s_wait_tensorcnt(2);  // current pair complete
      } else {
        __builtin_amdgcn_s_wait_tensorcnt(0);
      }
    }
    __syncthreads();
#else
    __syncthreads();
    #pragma unroll
    for (int it = 0; it < 2; ++it) {          // 512 chunks of 8 halves / matrix
      int idx = it * 256 + tid;
      int r  = idx >> 2;
      int c8 = (idx & 3) * 8;
      *(v8h*)&sA[0][r * LD + c8] = *(const v8h*)(Arow + (long)r * K + kk * 32 + c8);
      *(v8h*)&sB[0][r * LD + c8] = *(const v8h*)(Brow + (long)r * K + kk * 32 + c8);
    }
    __syncthreads();
#endif
    const _Float16* tA = &sA[USE_TDM ? buf : 0][0];
    const _Float16* tB = &sB[USE_TDM ? buf : 0][0];

    // A frags: lane<16 -> K{0..7,16..23}; lane>=16 -> K{8..15,24..31}
    v16h afrag[2];
    #pragma unroll
    for (int tm = 0; tm < 2; ++tm) {
      const _Float16* p = &tA[(wm*32 + tm*16 + r16) * LD + khalf * 8];
      V16u u; u.h[0] = *(const v8h*)p; u.h[1] = *(const v8h*)(p + 16);
      afrag[tm] = u.v;
    }
    // B frags: lane<16 -> K{0..7,16..23}; lane>=16 -> K{8..15,24..31}, column N = lane%16
    v16h bfrag[4];
    #pragma unroll
    for (int tn = 0; tn < 4; ++tn) {
      const _Float16* p = &tB[(wn*64 + tn*16 + r16) * LD + khalf * 8];
      V16u u; u.h[0] = *(const v8h*)p; u.h[1] = *(const v8h*)(p + 16);
      bfrag[tn] = u.v;
    }
    #pragma unroll
    for (int tm = 0; tm < 2; ++tm)
      #pragma unroll
      for (int tn = 0; tn < 4; ++tn)
        acc[tm][tn] = __builtin_amdgcn_wmma_f32_16x16x32_f16(
            false, afrag[tm], false, bfrag[tn], (short)0, acc[tm][tn], false, false);
    __syncthreads();
  }

  // Epilogue: lanes 0-15: VGPR j -> M=j; lanes 16-31: M=j+8
  const int g = lane >> 4;
  #pragma unroll
  for (int tm = 0; tm < 2; ++tm) {
    #pragma unroll
    for (int tn = 0; tn < 4; ++tn) {
      long col = colBase + wn*64 + tn*16 + r16;
      float bv = HAS_BIAS ? bias[col] : 0.0f;
      #pragma unroll
      for (int j = 0; j < 8; ++j) {
        long row = rowBase + wm*32 + tm*16 + g*8 + j;
        Cmat[row * N + col] = acc[tm][tn][j] + bv;
      }
    }
  }
}

// ---------------------------------------------------------------------------
// fp32 -> f16 conversion (4 elements / thread), n multiple of 1024
// ---------------------------------------------------------------------------
__global__ void cvt_f32_f16(const float* __restrict__ src, _Float16* __restrict__ dst)
{
  long i = (long)blockIdx.x * 256 + threadIdx.x;
  float4 f = ((const float4*)src)[i];
  v4h h; h[0] = (_Float16)f.x; h[1] = (_Float16)f.y; h[2] = (_Float16)f.z; h[3] = (_Float16)f.w;
  ((v4h*)dst)[i] = h;
}

// ---------------------------------------------------------------------------
// Agent token pooling: mean over 8x8 blocks of q (7x7 agent grid)
// ---------------------------------------------------------------------------
__global__ void pool_agents(const float* __restrict__ q, float* __restrict__ at)
{
  int b = blockIdx.x / AGENT;
  int a = blockIdx.x % AGENT;
  int py = a / 7, px = a % 7;
  for (int c = threadIdx.x; c < CH; c += 256) {
    float s = 0.f;
    for (int dy = 0; dy < 8; ++dy)
      for (int dx = 0; dx < 8; ++dx) {
        int i = (py*8 + dy) * 56 + px*8 + dx;
        s += q[((long)(b*NTOK + i)) * CH + c];
      }
    at[((long)(b*AGENT + a)) * CH + c] = s * (1.0f/64.0f);
  }
}

// ---------------------------------------------------------------------------
// Bilinear 7x7 -> 56x56 (half-pixel centers, edge clamp)
// ---------------------------------------------------------------------------
__device__ __forceinline__ float bilin7(const float* __restrict__ p, int hi, int wi)
{
  float yc = (hi + 0.5f) * 0.125f - 0.5f;
  float xc = (wi + 0.5f) * 0.125f - 0.5f;
  float y0f = floorf(yc), x0f = floorf(xc);
  float fy = yc - y0f, fx = xc - x0f;
  int y0 = (int)y0f, x0 = (int)x0f;
  int y0c = min(max(y0, 0), 6),   y1c = min(max(y0 + 1, 0), 6);
  int x0c = min(max(x0, 0), 6),   x1c = min(max(x0 + 1, 0), 6);
  float v00 = p[y0c*7 + x0c], v01 = p[y0c*7 + x1c];
  float v10 = p[y1c*7 + x0c], v11 = p[y1c*7 + x1c];
  return v00*(1.f-fy)*(1.f-fx) + v01*(1.f-fy)*fx + v10*fy*(1.f-fx) + v11*fy*fx;
}

__global__ void make_pb(const float* __restrict__ an, const float* __restrict__ ahb,
                        const float* __restrict__ awb, float* __restrict__ pb)
{
  long id = (long)blockIdx.x * 256 + threadIdx.x;   // heads*agent*ntok
  int i  = (int)(id % NTOK);
  int ha = (int)(id / NTOK);
  int hi = i / 56, wi = i % 56;
  float bval = bilin7(an + (long)ha * 49, hi, wi);
  pb[id] = bval + ahb[(long)ha*56 + hi] + awb[(long)ha*56 + wi];
}

__global__ void make_ab(const float* __restrict__ na, const float* __restrict__ hab,
                        const float* __restrict__ wab, float* __restrict__ ab)
{
  long id = (long)blockIdx.x * 256 + threadIdx.x;   // heads*agent*ntok
  int i  = (int)(id % NTOK);
  int ha = (int)(id / NTOK);
  int a = ha % AGENT, h = ha / AGENT;
  int hi = i / 56, wi = i % 56;
  float bval = bilin7(na + (long)ha * 49, hi, wi);
  ab[((long)(h*NTOK + i)) * AGENT + a] =
      bval + hab[((long)h*56 + hi) * AGENT + a] + wab[((long)h*56 + wi) * AGENT + a];
}

// ---------------------------------------------------------------------------
// Agent-side attention: block per (b,h,a). logits over n=3136 in LDS,
// softmax, then agent_v[b,h,a,:] = attn @ v_head
// ---------------------------------------------------------------------------
__global__ __launch_bounds__(256) void agent_attn_k(
    const float* __restrict__ at, const float* __restrict__ k,
    const float* __restrict__ v,  const float* __restrict__ pb,
    float* __restrict__ av)
{
  __shared__ __align__(16) float s_at[HD];
  __shared__ float s_l[NTOK];
  __shared__ float s_red[256];
  __shared__ float s_part[4][HD];
  int b   = blockIdx.x / (HEADS * AGENT);
  int rem = blockIdx.x % (HEADS * AGENT);
  int h = rem / AGENT, a = rem % AGENT;
  int tid = threadIdx.x;

  if (tid < HD) s_at[tid] = at[((long)(b*AGENT + a)) * CH + h*HD + tid] * SCALE;
  __syncthreads();

  const float* pbrow = pb + ((long)(h*AGENT + a)) * NTOK;
  for (int i = tid; i < NTOK; i += 256) {
    const float4* kp = (const float4*)(k + ((long)(b*NTOK + i)) * CH + h*HD);
    float d = 0.f;
    #pragma unroll
    for (int j = 0; j < 16; ++j) {
      float4 kv4 = kp[j];
      float4 aj  = ((const float4*)s_at)[j];
      d += kv4.x*aj.x + kv4.y*aj.y + kv4.z*aj.z + kv4.w*aj.w;
    }
    s_l[i] = d + pbrow[i];
  }
  __syncthreads();

  float m = -1e30f;
  for (int i = tid; i < NTOK; i += 256) m = fmaxf(m, s_l[i]);
  s_red[tid] = m; __syncthreads();
  for (int s = 128; s > 0; s >>= 1) {
    if (tid < s) s_red[tid] = fmaxf(s_red[tid], s_red[tid + s]);
    __syncthreads();
  }
  m = s_red[0]; __syncthreads();

  float sum = 0.f;
  for (int i = tid; i < NTOK; i += 256) { float e = __expf(s_l[i] - m); s_l[i] = e; sum += e; }
  s_red[tid] = sum; __syncthreads();
  for (int s = 128; s > 0; s >>= 1) {
    if (tid < s) s_red[tid] += s_red[tid + s];
    __syncthreads();
  }
  float inv = 1.0f / s_red[0];
  __syncthreads();

  int d = tid & 63, g = tid >> 6;
  float acc = 0.f;
  for (int i = g * 784; i < (g + 1) * 784; ++i)
    acc += s_l[i] * v[((long)(b*NTOK + i)) * CH + h*HD + d];
  s_part[g][d] = acc;
  __syncthreads();
  if (tid < HD) {
    float r = (s_part[0][tid] + s_part[1][tid] + s_part[2][tid] + s_part[3][tid]) * inv;
    av[((long)((b*HEADS + h)*AGENT + a)) * HD + tid] = r;
  }
}

// ---------------------------------------------------------------------------
// Query-side attention: block per (b,h, 8 pixels); one wave32 per pixel.
// 49-wide softmax via wave shuffles; agent tokens / agent_v in LDS.
// ---------------------------------------------------------------------------
__global__ __launch_bounds__(256) void q_attn_k(
    const float* __restrict__ q,  const float* __restrict__ at,
    const float* __restrict__ av, const float* __restrict__ ab,
    float* __restrict__ attnout)
{
  __shared__ float s_ah[AGENT * HD];
  __shared__ float s_av[AGENT * HD];
  __shared__ float s_q[8][HD];
  int bh   = blockIdx.x / 392;      // 392 = NTOK/8
  int iblk = blockIdx.x % 392;
  int b = bh / HEADS, h = bh % HEADS;
  int tid = threadIdx.x;

  for (int idx = tid; idx < AGENT * HD; idx += 256) {
    int a = idx >> 6, d = idx & 63;
    s_ah[idx] = at[((long)(b*AGENT + a)) * CH + h*HD + d];
    s_av[idx] = av[((long)((b*HEADS + h)*AGENT + a)) * HD + d];
  }
  int wid = tid >> 5, lane = tid & 31;
  int i = iblk * 8 + wid;
  long qbase = ((long)(b*NTOK + i)) * CH + h*HD;
  s_q[wid][lane]      = q[qbase + lane]      * SCALE;
  s_q[wid][lane + 32] = q[qbase + lane + 32] * SCALE;
  __syncthreads();

  const float* abrow = ab + ((long)(h*NTOK + i)) * AGENT;
  int a0 = lane, a1 = lane + 32;       // a1 valid for lanes 0..16 (49 agents)
  float l0 = 0.f, l1 = 0.f;
  #pragma unroll
  for (int d = 0; d < HD; ++d) {
    float qd = s_q[wid][d];
    l0 += qd * s_ah[a0*HD + d];
    if (lane < 17) l1 += qd * s_ah[a1*HD + d];
  }
  l0 += abrow[a0];
  l1 = (lane < 17) ? (l1 + abrow[a1]) : -1e30f;

  float m = fmaxf(l0, l1);
  for (int off = 16; off > 0; off >>= 1) m = fmaxf(m, __shfl_xor(m, off, 32));
  float e0 = __expf(l0 - m);
  float e1 = (lane < 17) ? __expf(l1 - m) : 0.f;
  float s = e0 + e1;
  for (int off = 16; off > 0; off >>= 1) s += __shfl_xor(s, off, 32);
  float inv = 1.0f / s;
  float p0 = e0 * inv, p1 = e1 * inv;

  float o0 = 0.f, o1 = 0.f;
  #pragma unroll
  for (int a = 0; a < AGENT; ++a) {
    float pa = (a < 32) ? __shfl(p0, a, 32) : __shfl(p1, a - 32, 32);
    o0 += pa * s_av[a*HD + lane];
    o1 += pa * s_av[a*HD + lane + 32];
  }
  attnout[qbase + lane]      = o0;
  attnout[qbase + lane + 32] = o1;
}

// ---------------------------------------------------------------------------
// Depthwise 3x3 conv on v + attn, emits f16 input for the proj GEMM
// ---------------------------------------------------------------------------
__global__ void dwc_add(const float* __restrict__ v, const float* __restrict__ w,
                        const float* __restrict__ bias, const float* __restrict__ attnin,
                        _Float16* __restrict__ attnh)
{
  long gid = (long)blockIdx.x * 256 + threadIdx.x;
  int c = (int)(gid & (CH - 1));
  long pix = gid >> 9;
  int i = (int)(pix % NTOK);
  int b = (int)(pix / NTOK);
  int hh = i / 56, ww = i % 56;
  float acc = bias[c];
  #pragma unroll
  for (int dy = 0; dy < 3; ++dy) {
    int hs = hh + dy - 1;
    if (hs < 0 || hs >= 56) continue;
    #pragma unroll
    for (int dx = 0; dx < 3; ++dx) {
      int ws = ww + dx - 1;
      if (ws < 0 || ws >= 56) continue;
      acc += v[((long)(b*NTOK + hs*56 + ws)) * CH + c] * w[c*9 + dy*3 + dx];
    }
  }
  attnh[gid] = (_Float16)(attnin[gid] + acc);
}

// ---------------------------------------------------------------------------
extern "C" void kernel_launch(void* const* d_in, const int* in_sizes, int n_in,
                              void* d_out, int out_size, void* d_ws, size_t ws_size,
                              hipStream_t stream)
{
  const float* x      = (const float*)d_in[0];
  const float* q_w    = (const float*)d_in[3];
  const float* kv_w   = (const float*)d_in[4];
  const float* proj_w = (const float*)d_in[5];
  const float* proj_b = (const float*)d_in[6];
  const float* dwc_w  = (const float*)d_in[7];
  const float* dwc_b  = (const float*)d_in[8];
  const float* an_b   = (const float*)d_in[9];
  const float* na_b   = (const float*)d_in[10];
  const float* ah_b   = (const float*)d_in[11];
  const float* aw_b   = (const float*)d_in[12];
  const float* ha_b   = (const float*)d_in[13];
  const float* wa_b   = (const float*)d_in[14];
  float* out = (float*)d_out;

  float* ws = (float*)d_ws;
  const size_t BNC = (size_t)MTOT * CH;               // 25,690,112
  float* q       = ws;
  float* k       = q + BNC;
  float* v       = k + BNC;
  float* agent_t = v + BNC;
  float* agent_v = agent_t + (size_t)BATCH * AGENT * CH;
  float* pb      = agent_v + (size_t)BATCH * AGENT * CH;
  float* ab      = pb + (size_t)HEADS * AGENT * NTOK;
  float* attn    = k;                                  // k is dead after agent_attn_k
  _Float16* xh   = (_Float16*)(ab + (size_t)HEADS * AGENT * NTOK);
  _Float16* qwh  = xh + BNC;                           // 512*512
  _Float16* kvwh = qwh + (size_t)CH * CH;              // 1024*512
  _Float16* pwh  = kvwh + (size_t)2 * CH * CH;         // 512*512
  _Float16* ath  = pwh + (size_t)CH * CH;              // attn in f16, BNC halves

  // fp32 -> f16 copies for the WMMA data path
  cvt_f32_f16<<<(int)(BNC / 1024), 256, 0, stream>>>(x, xh);
  cvt_f32_f16<<<(CH*CH) / 1024, 256, 0, stream>>>(q_w, qwh);
  cvt_f32_f16<<<(2*CH*CH) / 1024, 256, 0, stream>>>(kv_w, kvwh);
  cvt_f32_f16<<<(CH*CH) / 1024, 256, 0, stream>>>(proj_w, pwh);

  dim3 gg(MTOT / 128, CH / 128);   // 392 x 4
  gemm512_wmma<false><<<gg, 256, 0, stream>>>(xh, qwh,              q, nullptr);
  gemm512_wmma<false><<<gg, 256, 0, stream>>>(xh, kvwh,             k, nullptr);
  gemm512_wmma<false><<<gg, 256, 0, stream>>>(xh, kvwh + CH*CH,     v, nullptr);

  pool_agents<<<BATCH * AGENT, 256, 0, stream>>>(q, agent_t);

  int nbias = HEADS * AGENT * NTOK;                   // 1,229,312
  make_pb<<<nbias / 256, 256, 0, stream>>>(an_b, ah_b, aw_b, pb);
  make_ab<<<nbias / 256, 256, 0, stream>>>(na_b, ha_b, wa_b, ab);

  agent_attn_k<<<BATCH * HEADS * AGENT, 256, 0, stream>>>(agent_t, k, v, pb, agent_v);
  q_attn_k<<<BATCH * HEADS * (NTOK / 8), 256, 0, stream>>>(q, agent_t, agent_v, ab, attn);

  dwc_add<<<(int)(BNC / 256), 256, 0, stream>>>(v, dwc_w, dwc_b, attn, ath);

  // out = attn @ proj_w^T + proj_b (flat reinterpretation == final reshape)
  gemm512_wmma<true><<<gg, 256, 0, stream>>>(ath, pwh, out, proj_b);
}
